// SparseConvAttnBlock_46256797778305
// MI455X (gfx1250) — compile-verified, matrix operand order checked by reading, not run on hardware
//
#include <hip/hip_runtime.h>

// ---------------------------------------------------------------------------
// Types for CDNA5 WMMA (wave32, 16x16x32 f16 -> f32)
// ---------------------------------------------------------------------------
typedef __attribute__((ext_vector_type(16))) _Float16 v16h;
typedef __attribute__((ext_vector_type(8)))  float    v8f;
typedef __attribute__((ext_vector_type(8)))  _Float16 h8;
typedef int i32x4v __attribute__((vector_size(16)));

#define WMMA_F16(A, B, C) \
  __builtin_amdgcn_wmma_f32_16x16x32_f16(false, (A), false, (B), (short)0, (C), false, false)

// Per ISA 7.12.2 (16-bit A 16x32): lane<16 holds M=lane, K={kb..kb+7, kb+16..kb+23}
// with kb = (lane>>4)*8. B operand (K x N) is loaded identically from B^T storage.
__device__ __forceinline__ v16h load_tile(const _Float16* p) {
  h8 lo = *(const h8*)(p);
  h8 hi = *(const h8*)(p + 16);
  v16h r;
#pragma unroll
  for (int j = 0; j < 8; ++j) { r[j] = lo[j]; r[j + 8] = hi[j]; }
  return r;
}

// Strided gather variant (for V as B-operand: rows are keys, stride = row pitch).
__device__ __forceinline__ v16h load_tile_col(const _Float16* p, int ld) {
  v16h r;
#pragma unroll
  for (int j = 0; j < 8; ++j) { r[j] = p[j * ld]; r[j + 8] = p[(16 + j) * ld]; }
  return r;
}

// ---------------------------------------------------------------------------
// CDNA5 async global->LDS staging (GLOBAL_LOAD_ASYNC_TO_LDS_B128, ASYNCcnt).
// The builtin on this toolchain takes generic int4* pointers (per round-2
// diagnostic); codegen inserts the AS1/AS3 casts.  Falls back to plain LDS
// stores if the builtin is absent.
// ---------------------------------------------------------------------------
#if __has_builtin(__builtin_amdgcn_global_load_async_to_lds_b128)
#define HAVE_ASYNC_LDS 1
#endif

__device__ __forceinline__ void async_copy16(const _Float16* g, _Float16* l) {
#ifdef HAVE_ASYNC_LDS
  __builtin_amdgcn_global_load_async_to_lds_b128((i32x4v*)(g), (i32x4v*)(l), 0, 0);
#else
  *(h8*)l = *(const h8*)g;
#endif
}

__device__ __forceinline__ void async_wait_all() {
#ifdef HAVE_ASYNC_LDS
#if __has_builtin(__builtin_amdgcn_s_wait_asynccnt)
  __builtin_amdgcn_s_wait_asynccnt(0);
#else
  asm volatile("s_wait_asynccnt 0" ::: "memory");
#endif
#endif
}

// ---------------------------------------------------------------------------
// Constants
// ---------------------------------------------------------------------------
#define RR 32
#define CC_CH 256
#define NH 8
#define HD 32
#define NTOK (RR * RR * RR)   // 32768
#define KC 64                 // K chunk staged in LDS per GEMM iteration

// ---------------------------------------------------------------------------
// Weight conversion: out[b][o][i] = (f16) in[b][i][o]   (B^T storage for WMMA)
// ---------------------------------------------------------------------------
__global__ void transpose_to_f16(const float* __restrict__ in, _Float16* __restrict__ out,
                                 int I, int O, int nbat) {
  size_t total = (size_t)nbat * I * O;
  size_t id = (size_t)blockIdx.x * 256 + threadIdx.x;
  if (id >= total) return;
  size_t mat = id / ((size_t)I * O);
  size_t r = id % ((size_t)I * O);
  int o = (int)(r / I), i = (int)(r % I);
  out[mat * I * O + (size_t)o * I + i] = (_Float16)in[mat * I * O + (size_t)i * O + o];
}

__global__ void to_f16_kernel(const float* __restrict__ in, _Float16* __restrict__ out, int n) {
  int i = blockIdx.x * 256 + threadIdx.x;
  if (i < n) out[i] = (_Float16)in[i];
}

// ---------------------------------------------------------------------------
// adaLN modulation: mod[1536] = silu(t_emb) @ ada_w + ada_b
// ---------------------------------------------------------------------------
__global__ void mod_kernel(const float* __restrict__ t_emb, const float* __restrict__ ada_w,
                           const float* __restrict__ ada_b, float* __restrict__ mod) {
  int j = blockIdx.x * 256 + threadIdx.x;
  if (j >= 6 * CC_CH) return;
  float acc = ada_b[j];
  for (int i = 0; i < CC_CH; ++i) {
    float t = t_emb[i];
    float s = t / (1.f + __expf(-t));
    acc += s * ada_w[i * (6 * CC_CH) + j];
  }
  mod[j] = acc;
}

// ---------------------------------------------------------------------------
// LayerNorm variants -> f16 output.  mode 0: silu(ln*g+b)  mode 1: ln*(1+s)+t  mode 2: ln
// One block (256 threads) per row.
// ---------------------------------------------------------------------------
__global__ void ln_kernel(const float* __restrict__ x, const float* __restrict__ p0,
                          const float* __restrict__ p1, _Float16* __restrict__ out, int mode) {
  __shared__ float red[256];
  int n = blockIdx.x, c = threadIdx.x;
  float v = x[(size_t)n * CC_CH + c];
  red[c] = v;
  __syncthreads();
  for (int s = 128; s > 0; s >>= 1) { if (c < s) red[c] += red[c + s]; __syncthreads(); }
  float mu = red[0] * (1.f / CC_CH);
  __syncthreads();
  float xc = v - mu;
  red[c] = xc * xc;
  __syncthreads();
  for (int s = 128; s > 0; s >>= 1) { if (c < s) red[c] += red[c + s]; __syncthreads(); }
  float var = red[0] * (1.f / CC_CH);
  float y = xc * rsqrtf(var + 1e-6f);
  if (mode == 0) { y = y * p0[c] + p1[c]; y = y / (1.f + __expf(-y)); }
  else if (mode == 1) { y = y * (1.f + p0[c]) + p1[c]; }
  out[(size_t)n * CC_CH + c] = (_Float16)y;
}

// ---------------------------------------------------------------------------
// Generic WMMA GEMM with async-LDS B staging and 4x register blocking.
// Block (256 thr / 8 waves) computes a 128(M) x 64(N) panel: wave w owns
// M-tile w and 4 N-tiles; the 64xKC B panel is staged into LDS once per chunk
// (double-buffered, ASYNCcnt) and shared by all 8 waves.
// MODE 0: f16 store; 1: f16 gelu; 2: f32 res + v; 3: f32 res + gate[col]*v
// ---------------------------------------------------------------------------
template <int MODE>
__global__ void __launch_bounds__(256) gemm_wmma(
    const _Float16* __restrict__ A, int lda,
    const _Float16* __restrict__ Bt,
    const float* __restrict__ bias,
    const float* __restrict__ res,
    const float* __restrict__ gate,
    void* __restrict__ dst, int ldc,
    int M, int N, int K) {
  __shared__ alignas(16) _Float16 bs[2][64 * KC];
  int tid = threadIdx.x;
  int wave = tid >> 5, lane = tid & 31;
  int nb = N >> 6;
  int bm = blockIdx.x / nb, bn = blockIdx.x % nb;
  int n0 = bn << 6;
  int row = lane & 15, kb = (lane >> 4) << 3;
  int mb = (bm << 7) + (wave << 4);

  // staging: 512 x 16B units per chunk, 2 per thread
  int u0 = tid * 2, u1 = u0 + 1;
  int sr0 = u0 >> 3, sc0 = (u0 & 7) << 3;
  int sr1 = u1 >> 3, sc1 = (u1 & 7) << 3;
  auto stage = [&](int k0, int buf) {
    async_copy16(Bt + (size_t)(n0 + sr0) * K + k0 + sc0, &bs[buf][sr0 * KC + sc0]);
    async_copy16(Bt + (size_t)(n0 + sr1) * K + k0 + sc1, &bs[buf][sr1 * KC + sc1]);
  };

  v8f acc[4] = {};
  const _Float16* arow = A + (size_t)(mb + row) * lda + kb;
  int nc = K >> 6;
  int buf = 0;
  stage(0, 0);
  for (int c = 0; c < nc; ++c) {
    async_wait_all();
    __syncthreads();
    if (c + 1 < nc) stage((c + 1) << 6, buf ^ 1);
    int k0 = c << 6;
    __builtin_prefetch(arow + k0 + KC, 0, 1);
#pragma unroll
    for (int kk = 0; kk < KC; kk += 32) {
      v16h a = load_tile(arow + k0 + kk);
#pragma unroll
      for (int nt = 0; nt < 4; ++nt) {
        v16h b = load_tile(&bs[buf][(nt * 16 + row) * KC + kk + kb]);
        acc[nt] = WMMA_F16(a, b, acc[nt]);
      }
    }
    __syncthreads();
    buf ^= 1;
  }

  int mr = (lane >> 4) << 3;
#pragma unroll
  for (int nt = 0; nt < 4; ++nt) {
    int col = n0 + nt * 16 + row;
    float bv = bias ? bias[col] : 0.f;
#pragma unroll
    for (int r = 0; r < 8; ++r) {
      int m = mb + mr + r;
      float v = acc[nt][r] + bv;
      if (MODE == 0) {
        ((_Float16*)dst)[(size_t)m * ldc + col] = (_Float16)v;
      } else if (MODE == 1) {
        float u = 0.7978845608028654f * (v + 0.044715f * v * v * v);
        ((_Float16*)dst)[(size_t)m * ldc + col] = (_Float16)(0.5f * v * (1.f + tanhf(u)));
      } else if (MODE == 2) {
        ((float*)dst)[(size_t)m * ldc + col] = res[(size_t)m * ldc + col] + v;
      } else {
        ((float*)dst)[(size_t)m * ldc + col] = res[(size_t)m * ldc + col] + gate[col] * v;
      }
    }
  }
}

// ---------------------------------------------------------------------------
// 3x3x3 conv as implicit GEMM over 27 taps (K = 27*256), residual + bias.
// Wave computes 16(M) x 64(N): A tile loaded once per (tap, k-step) feeds 4
// WMMAs.  16 consecutive tokens share (ix,iy); z-boundary handled per lane.
// ---------------------------------------------------------------------------
__global__ void __launch_bounds__(256) conv_wmma(
    const _Float16* __restrict__ h16, const _Float16* __restrict__ convWt,
    const float* __restrict__ x, const float* __restrict__ conv_b,
    float* __restrict__ xout) {
  int wave = threadIdx.x >> 5, lane = threadIdx.x & 31;
  int task = blockIdx.x * 8 + wave;           // 2048 * 4 tasks
  int tm = task >> 2, tc = task & 3;
  int n0 = tm * 16;
  int co0 = tc * 64;
  int iz0 = n0 & 31, iy = (n0 >> 5) & 31, ix = n0 >> 10;
  int row = lane & 15, kb = (lane >> 4) << 3;
  int iz = iz0 + row;
  v8f acc[4] = {};
  for (int t = 0; t < 27; ++t) {
    int dx = t / 9 - 1, dy = (t / 3) % 3 - 1, dz = t % 3 - 1;
    int jx = ix + dx, jy = iy + dy, jz = iz + dz;
    bool valid = ((unsigned)jx < 32u) & ((unsigned)jy < 32u) & ((unsigned)jz < 32u);
    const _Float16* ap = h16 + ((size_t)((jx << 10) + (jy << 5) + jz)) * CC_CH + kb;
    const _Float16* bp = convWt + (size_t)t * 65536 + (size_t)(co0 + row) * CC_CH + kb;
    for (int k0 = 0; k0 < CC_CH; k0 += 32) {
      v16h a = {};
      if (valid) a = load_tile(ap + k0);
#pragma unroll
      for (int nt = 0; nt < 4; ++nt) {
        v16h b = load_tile(bp + (size_t)nt * 16 * CC_CH + k0);
        acc[nt] = WMMA_F16(a, b, acc[nt]);
      }
    }
  }
  int mr = (lane >> 4) << 3;
#pragma unroll
  for (int nt = 0; nt < 4; ++nt) {
    int col = co0 + nt * 16 + row;
#pragma unroll
    for (int r = 0; r < 8; ++r) {
      int m = n0 + mr + r;
      xout[(size_t)m * CC_CH + col] = x[(size_t)m * CC_CH + col] + conv_b[col] + acc[nt][r];
    }
  }
}

// ---------------------------------------------------------------------------
// RoPE + windowize: read qkv (N x 768 f16), rotate q/k, write window-contiguous
// qw/kw/vw ([64 windows][512 tok][256 ch]).  One thread per (n, head, pair).
// ---------------------------------------------------------------------------
__global__ void rope_window_kernel(const _Float16* __restrict__ qkv,
                                   _Float16* __restrict__ qw, _Float16* __restrict__ kw,
                                   _Float16* __restrict__ vw) {
  int tid = blockIdx.x * 256 + threadIdx.x;       // N*H*16
  int j = tid & 15;
  int h = (tid >> 4) & 7;
  int n = tid >> 7;
  int iz = n & 31, iy = (n >> 5) & 31, ix = n >> 10;
  float ph = 0.f;
  if (j < 15) {
    int axis = j / 5, fi = j % 5;
    float coord = (axis == 0) ? (float)ix : (axis == 1) ? (float)iy : (float)iz;
    ph = coord * __powf(10000.f, -(float)fi * 0.2f);
  }
  float cs = __cosf(ph), sn = __sinf(ph);
  const _Float16* qp = qkv + (size_t)n * 768 + h * HD + 2 * j;
  const _Float16* kp = qp + 256;
  const _Float16* vp = qp + 512;
  float qr = (float)qp[0], qi = (float)qp[1];
  float kr = (float)kp[0], ki = (float)kp[1];
  int w = ((ix >> 3) * 4 + (iy >> 3)) * 4 + (iz >> 3);
  int q = ((ix & 7) * 8 + (iy & 7)) * 8 + (iz & 7);
  size_t o = ((size_t)(w * 512 + q)) * CC_CH + h * HD + 2 * j;
  qw[o]     = (_Float16)(qr * cs - qi * sn);
  qw[o + 1] = (_Float16)(qr * sn + qi * cs);
  kw[o]     = (_Float16)(kr * cs - ki * sn);
  kw[o + 1] = (_Float16)(kr * sn + ki * cs);
  vw[o]     = vp[0];
  vw[o + 1] = vp[1];
}

__device__ __forceinline__ int unwindow(int w, int q) {
  int wx = (w >> 4) & 3, wy = (w >> 2) & 3, wz = w & 3;
  int lx = (q >> 6) & 7, ly = (q >> 3) & 7, lz = q & 7;
  return ((wx * 8 + lx) << 10) + ((wy * 8 + ly) << 5) + (wz * 8 + lz);
}

// ---------------------------------------------------------------------------
// Windowed attention: one wave per (window, head, 16-query tile).
// Scores 16x512 in LDS (32KB) + f16 probs (16KB).  QK^T and PV via WMMA.
// ---------------------------------------------------------------------------
__global__ void __launch_bounds__(32) win_attn_kernel(
    const _Float16* __restrict__ qw, const _Float16* __restrict__ kw,
    const _Float16* __restrict__ vw, _Float16* __restrict__ aout) {
  __shared__ alignas(16) float    sc[16 * 512];
  __shared__ alignas(16) _Float16 pr[16 * 512];
  int lane = threadIdx.x;
  int b = blockIdx.x;               // 64*8*32
  int w = b >> 8, h = (b >> 5) & 7, qt = b & 31;
  int row = lane & 15, kb = (lane >> 4) << 3;
  int col = row, mr = (lane >> 4) << 3;
  const size_t wbase = (size_t)w * 512 * CC_CH;
  const float scale = 0.17677669529663687f;  // 1/sqrt(32)

  v16h a = load_tile(qw + wbase + (size_t)(qt * 16 + row) * CC_CH + h * HD + kb);
  for (int kt = 0; kt < 32; ++kt) {
    v16h bk = load_tile(kw + wbase + (size_t)(kt * 16 + row) * CC_CH + h * HD + kb);
    v8f c = {};
    c = WMMA_F16(a, bk, c);
#pragma unroll
    for (int r = 0; r < 8; ++r) sc[(mr + r) * 512 + kt * 16 + col] = c[r] * scale;
  }
  __syncthreads();
  if (lane < 16) {
    float mx = -1e30f;
    for (int kk = 0; kk < 512; ++kk) mx = fmaxf(mx, sc[lane * 512 + kk]);
    float s = 0.f;
    for (int kk = 0; kk < 512; ++kk) {
      float e = __expf(sc[lane * 512 + kk] - mx);
      s += e;
      pr[lane * 512 + kk] = (_Float16)e;
    }
    float rinv = 1.f / s;
    for (int kk = 0; kk < 512; ++kk)
      pr[lane * 512 + kk] = (_Float16)((float)pr[lane * 512 + kk] * rinv);
  }
  __syncthreads();
  v8f o0 = {}, o1 = {};
  for (int pt = 0; pt < 16; ++pt) {
    v16h ap = load_tile(pr + row * 512 + pt * 32 + kb);
    const _Float16* v0 = vw + wbase + (size_t)(pt * 32 + kb) * CC_CH + h * HD + col;
    v16h b0 = load_tile_col(v0, CC_CH);
    v16h b1 = load_tile_col(v0 + 16, CC_CH);
    o0 = WMMA_F16(ap, b0, o0);
    o1 = WMMA_F16(ap, b1, o1);
  }
#pragma unroll
  for (int r = 0; r < 8; ++r) {
    int n = unwindow(w, qt * 16 + mr + r);
    aout[(size_t)n * CC_CH + h * HD + col]      = (_Float16)o0[r];
    aout[(size_t)n * CC_CH + h * HD + 16 + col] = (_Float16)o1[r];
  }
}

// ---------------------------------------------------------------------------
// Cross attention vs L=256 cond tokens; kv16 = [L][512] (ck | cv).
// Two waves per block, each handling 16 query rows of one head.
// ---------------------------------------------------------------------------
__global__ void __launch_bounds__(64) cross_attn_kernel(
    const _Float16* __restrict__ q16, const _Float16* __restrict__ kv16,
    _Float16* __restrict__ o16) {
  __shared__ alignas(16) float    sc[2][16 * 256];
  __shared__ alignas(16) _Float16 pr[2][16 * 256];
  int wave = threadIdx.x >> 5, lane = threadIdx.x & 31;
  int id = blockIdx.x * 2 + wave;       // 8 * 2048
  int h = id >> 11, mt = id & 2047;
  int row = lane & 15, kb = (lane >> 4) << 3;
  int col = row, mr = (lane >> 4) << 3;
  float* S = sc[wave];
  _Float16* P = pr[wave];
  const float scale = 0.17677669529663687f;

  v16h a = load_tile(q16 + (size_t)(mt * 16 + row) * CC_CH + h * HD + kb);
  for (int kt = 0; kt < 16; ++kt) {
    v16h bk = load_tile(kv16 + (size_t)(kt * 16 + row) * 512 + h * HD + kb);
    v8f c = {};
    c = WMMA_F16(a, bk, c);
#pragma unroll
    for (int r = 0; r < 8; ++r) S[(mr + r) * 256 + kt * 16 + col] = c[r] * scale;
  }
  __syncthreads();
  if (lane < 16) {
    float mx = -1e30f;
    for (int kk = 0; kk < 256; ++kk) mx = fmaxf(mx, S[lane * 256 + kk]);
    float s = 0.f;
    for (int kk = 0; kk < 256; ++kk) {
      float e = __expf(S[lane * 256 + kk] - mx);
      s += e;
      P[lane * 256 + kk] = (_Float16)e;
    }
    float rinv = 1.f / s;
    for (int kk = 0; kk < 256; ++kk)
      P[lane * 256 + kk] = (_Float16)((float)P[lane * 256 + kk] * rinv);
  }
  __syncthreads();
  v8f o0 = {}, o1 = {};
  for (int pt = 0; pt < 8; ++pt) {
    v16h ap = load_tile(P + row * 256 + pt * 32 + kb);
    const _Float16* v0 = kv16 + (size_t)(pt * 32 + kb) * 512 + 256 + h * HD + col;
    v16h b0 = load_tile_col(v0, 512);
    v16h b1 = load_tile_col(v0 + 16, 512);
    o0 = WMMA_F16(ap, b0, o0);
    o1 = WMMA_F16(ap, b1, o1);
  }
#pragma unroll
  for (int r = 0; r < 8; ++r) {
    int n = mt * 16 + mr + r;
    o16[(size_t)n * CC_CH + h * HD + col]      = (_Float16)o0[r];
    o16[(size_t)n * CC_CH + h * HD + 16 + col] = (_Float16)o1[r];
  }
}

// ---------------------------------------------------------------------------
// Host-side orchestration
// ---------------------------------------------------------------------------
extern "C" void kernel_launch(void* const* d_in, const int* in_sizes, int n_in,
                              void* d_out, int out_size, void* d_ws, size_t ws_size,
                              hipStream_t stream) {
  const float* x      = (const float*)d_in[0];
  const float* t_emb  = (const float*)d_in[1];
  const float* cond   = (const float*)d_in[2];
  const float* ln_g   = (const float*)d_in[3];
  const float* ln_b   = (const float*)d_in[4];
  const float* conv_w = (const float*)d_in[5];
  const float* conv_b = (const float*)d_in[6];
  const float* ada_w  = (const float*)d_in[7];
  const float* ada_b  = (const float*)d_in[8];
  const float* qkv_w  = (const float*)d_in[9];
  const float* qkv_b  = (const float*)d_in[10];
  const float* so_w   = (const float*)d_in[11];
  const float* so_b   = (const float*)d_in[12];
  const float* cq_w   = (const float*)d_in[13];
  const float* cq_b   = (const float*)d_in[14];
  const float* ckv_w  = (const float*)d_in[15];
  const float* ckv_b  = (const float*)d_in[16];
  const float* co_w   = (const float*)d_in[17];
  const float* co_b   = (const float*)d_in[18];
  const float* m1_w   = (const float*)d_in[19];
  const float* m1_b   = (const float*)d_in[20];
  const float* m2_w   = (const float*)d_in[21];
  const float* m2_b   = (const float*)d_in[22];

  char* p = (char*)d_ws;
  auto carve = [&](size_t bytes) -> char* {
    char* r = p;
    p += (bytes + 255) & ~(size_t)255;
    return r;
  };
  float*     mod    = (float*)    carve(1536 * 4);
  _Float16*  convWt = (_Float16*) carve((size_t)27 * 65536 * 2);
  _Float16*  qkvWt  = (_Float16*) carve((size_t)768 * 256 * 2);
  _Float16*  soWt   = (_Float16*) carve((size_t)256 * 256 * 2);
  _Float16*  cqWt   = (_Float16*) carve((size_t)256 * 256 * 2);
  _Float16*  ckvWt  = (_Float16*) carve((size_t)512 * 1024 * 2);
  _Float16*  coWt   = (_Float16*) carve((size_t)256 * 256 * 2);
  _Float16*  m1Wt   = (_Float16*) carve((size_t)1024 * 256 * 2);
  _Float16*  m2Wt   = (_Float16*) carve((size_t)256 * 1024 * 2);
  _Float16*  cond16 = (_Float16*) carve((size_t)256 * 1024 * 2);
  _Float16*  kv16   = (_Float16*) carve((size_t)256 * 512 * 2);
  _Float16*  h16    = (_Float16*) carve((size_t)NTOK * 256 * 2);
  _Float16*  a16    = (_Float16*) carve((size_t)NTOK * 256 * 2);
  float*     xA     = (float*)    carve((size_t)NTOK * 256 * 4);
  float*     xB     = (float*)    carve((size_t)NTOK * 256 * 4);
  _Float16*  qw     = (_Float16*) carve((size_t)NTOK * 256 * 2);
  _Float16*  kw     = (_Float16*) carve((size_t)NTOK * 256 * 2);
  _Float16*  vw     = (_Float16*) carve((size_t)NTOK * 256 * 2);
  _Float16*  big    = (_Float16*) carve((size_t)NTOK * 1024 * 2);  // qkv16 then mlp16
  _Float16*  qkv16  = big;
  _Float16*  mlp16  = big;

  // --- weight prep (f16, transposed to [N][K]) ---
  transpose_to_f16<<<6912, 256, 0, stream>>>(conv_w, convWt, 256, 256, 27);
  transpose_to_f16<<<768,  256, 0, stream>>>(qkv_w,  qkvWt, 256, 768, 1);
  transpose_to_f16<<<256,  256, 0, stream>>>(so_w,   soWt,  256, 256, 1);
  transpose_to_f16<<<256,  256, 0, stream>>>(cq_w,   cqWt,  256, 256, 1);
  transpose_to_f16<<<2048, 256, 0, stream>>>(ckv_w,  ckvWt, 1024, 512, 1);
  transpose_to_f16<<<256,  256, 0, stream>>>(co_w,   coWt,  256, 256, 1);
  transpose_to_f16<<<1024, 256, 0, stream>>>(m1_w,   m1Wt,  256, 1024, 1);
  transpose_to_f16<<<1024, 256, 0, stream>>>(m2_w,   m2Wt,  1024, 256, 1);
  to_f16_kernel<<<1024, 256, 0, stream>>>(cond, cond16, 256 * 1024);
  mod_kernel<<<6, 256, 0, stream>>>(t_emb, ada_w, ada_b, mod);

  // --- conv branch: h = silu(ln(x)*g+b); xA = x + conv3d(h) + conv_b ---
  ln_kernel<<<NTOK, 256, 0, stream>>>(x, ln_g, ln_b, h16, 0);
  conv_wmma<<<1024, 256, 0, stream>>>(h16, convWt, x, conv_b, xA);

  // --- window attention branch ---
  ln_kernel<<<NTOK, 256, 0, stream>>>(xA, mod + 256, mod + 0, h16, 1);  // sc_m, sh_m
  gemm_wmma<0><<<3072, 256, 0, stream>>>(h16, 256, qkvWt, qkv_b, nullptr, nullptr,
                                         (void*)qkv16, 768, NTOK, 768, 256);
  rope_window_kernel<<<16384, 256, 0, stream>>>(qkv16, qw, kw, vw);
  win_attn_kernel<<<16384, 32, 0, stream>>>(qw, kw, vw, a16);
  gemm_wmma<3><<<1024, 256, 0, stream>>>(a16, 256, soWt, so_b, xA, mod + 512,
                                         (void*)xB, 256, NTOK, 256, 256);  // g_m gated

  // --- cross attention branch ---
  ln_kernel<<<NTOK, 256, 0, stream>>>(xB, nullptr, nullptr, h16, 2);
  gemm_wmma<0><<<1024, 256, 0, stream>>>(h16, 256, cqWt, cq_b, nullptr, nullptr,
                                         (void*)a16, 256, NTOK, 256, 256);
  gemm_wmma<0><<<16, 256, 0, stream>>>(cond16, 1024, ckvWt, ckv_b, nullptr, nullptr,
                                       (void*)kv16, 512, 256, 512, 1024);
  cross_attn_kernel<<<8192, 64, 0, stream>>>(a16, kv16, h16);
  gemm_wmma<2><<<1024, 256, 0, stream>>>(h16, 256, coWt, co_b, xB, nullptr,
                                         (void*)xA, 256, NTOK, 256, 256);

  // --- MLP branch ---
  ln_kernel<<<NTOK, 256, 0, stream>>>(xA, mod + 1024, mod + 768, h16, 1);  // sc_f, sh_f
  gemm_wmma<1><<<4096, 256, 0, stream>>>(h16, 256, m1Wt, m1_b, nullptr, nullptr,
                                         (void*)mlp16, 1024, NTOK, 1024, 256);
  gemm_wmma<3><<<1024, 256, 0, stream>>>(mlp16, 1024, m2Wt, m2_b, xA, mod + 1280,
                                         d_out, 256, NTOK, 256, 1024);  // g_f gated
}
